// DSConvMMCV_35699768165208
// MI455X (gfx1250) — matrix operand order repaired and checked by reading, not compiled, not run on hardware
//
#include <hip/hip_runtime.h>
#include <hip/hip_bf16.h>
#include <stdint.h>

#define Nb    4
#define Cin   128
#define Hh    96
#define Ww    96
#define HW    (Hh*Ww)        // 9216
#define NPIX  (Nb*HW)        // 36864
#define COUT  128
#define KT    9
#define KTOT  (KT*Cin)       // 1152
#define GN_G  8
#define GPC   16             // channels per group

typedef __attribute__((ext_vector_type(16))) __bf16 v16bf;
typedef __attribute__((ext_vector_type(8)))  float  v8f;

// ---- workspace layout (float units) ----
#define XT_OFF  0ull
#define Y_OFF   (XT_OFF + (size_t)NPIX*Cin)     // 4.7M floats
#define DX_OFF  (Y_OFF  + (size_t)NPIX*COUT)
#define DY_OFF  (DX_OFF + (size_t)NPIX)
#define MK_OFF  (DY_OFF + (size_t)NPIX)
#define ST_OFF  (MK_OFF + (size_t)NPIX)          // 128 floats: sum[32] sq[32] mu[32] rs[32]
#define BP_OFF  (ST_OFF + 128)                   // 73728 uint32 (bf16 B fragments)

// reference tap offset coefficients: oy = oyc[t]*dx, ox = oxc[t]*dy
__constant__ float c_oyc[9] = {0.f,0.f,0.f, 1.f,0.f,-1.f, 1.f,1.f,-1.f};
__constant__ float c_oxc[9] = {0.f,1.f,-1.f, 0.f,0.f,0.f, 1.f,-1.f,1.f};

// single-instruction packed f32->bf16 (CDNA5 VOP3: dst.lo=cvt(s0), dst.hi=cvt(s1))
__device__ __forceinline__ uint32_t pack_bf16x2(float lo, float hi) {
    uint32_t r;
    asm("v_cvt_pk_bf16_f32 %0, %1, %2" : "=v"(r) : "v"(lo), "v"(hi));
    return r;
}

__device__ __forceinline__ int iclamp(int v, int lo, int hi) {
    return v < lo ? lo : (v > hi ? hi : v);
}

// ---------------- kernel 0: zero GN stats ----------------
__global__ void k_zero(float* __restrict__ st) {
    if (threadIdx.x < 128) st[threadIdx.x] = 0.f;
}

// ---------------- kernel 1: NCHW -> NHWC ----------------
__global__ void k_transpose(const float* __restrict__ x, float* __restrict__ xt) {
    int pix = blockIdx.x;
    int c   = threadIdx.x;
    int n   = pix / HW;
    int hw  = pix % HW;
    xt[(size_t)pix * Cin + c] = x[((size_t)n * Cin + c) * HW + hw];
}

// ---------------- kernel 2: pack w_dcn into bf16 WMMA B fragments ----------------
// K order = tap*Cin + c. Fragment layout (32x16 bf16 B): lane holds col N=lane%16;
// lanes 0-15 hold K=0..15 (dword v -> K=2v,2v+1), lanes 16-31 hold K=16..31.
// Storage: bp[(((kc*8)+nt)*32 + lane)*8 + v]
__global__ void k_packb(const float* __restrict__ wd, uint32_t* __restrict__ bp) {
    int tid  = blockIdx.x * 256 + threadIdx.x;     // 73728 total
    int v    = tid & 7;
    int lane = (tid >> 3) & 31;
    int nt   = (tid >> 8) & 7;
    int kc   = tid >> 11;                          // 0..35
    int ncol = nt * 16 + (lane & 15);
    int half = lane >> 4;
    int k0   = kc * 32 + half * 16 + 2 * v;
    int k1   = k0 + 1;
    int tap0 = k0 / Cin, ch0 = k0 % Cin;
    int tap1 = k1 / Cin, ch1 = k1 % Cin;
    float f0 = wd[((size_t)ncol * Cin + ch0) * KT + tap0];
    float f1 = wd[((size_t)ncol * Cin + ch1) * KT + tap1];
    bp[tid] = pack_bf16x2(f0, f1);
}

// ---------------- kernel 3: head conv -> dx, dy, mask ----------------
__global__ void k_head(const float* __restrict__ xt, const float* __restrict__ wh,
                       const float* __restrict__ bh,
                       float* __restrict__ dxp, float* __restrict__ dyp,
                       float* __restrict__ mp) {
    __shared__ float w_s[3 * Cin * KT];            // reorg: w_s[t*384 + c*3 + o]
    for (int i = threadIdx.x; i < 3 * Cin * KT; i += 128) {
        int o = i % 3;
        int c = (i / 3) % Cin;
        int t = i / (3 * Cin);
        w_s[i] = wh[(size_t)o * Cin * KT + (size_t)c * KT + t];
    }
    __syncthreads();

    int pix = blockIdx.x * 128 + threadIdx.x;
    int n  = pix / HW;
    int hw = pix % HW;
    int h  = hw / Ww, w = hw % Ww;
    float a0 = bh[0], a1 = bh[1], a2 = bh[2];

    for (int t = 0; t < 9; ++t) {
        int hy = h - 1 + t / 3;
        int wx = w - 1 + t % 3;
        if (hy < 0 || hy >= Hh || wx < 0 || wx >= Ww) continue;
        const float4* p = (const float4*)(xt + ((size_t)n * HW + (size_t)hy * Ww + wx) * Cin);
        const float* wt = &w_s[t * 384];
        for (int c4 = 0; c4 < 32; ++c4) {
            float4 v = p[c4];
            const float* wc = wt + c4 * 12;
            a0 += v.x * wc[0] + v.y * wc[3] + v.z * wc[6] + v.w * wc[9];
            a1 += v.x * wc[1] + v.y * wc[4] + v.z * wc[7] + v.w * wc[10];
            a2 += v.x * wc[2] + v.y * wc[5] + v.z * wc[8] + v.w * wc[11];
        }
    }
    dxp[pix] = 0.25f * tanhf(a0);
    dyp[pix] = 0.25f * tanhf(a1);
    mp[pix]  = 1.f / (1.f + __expf(-a2));
}

// ---------------- kernel 4: deformable sampling + WMMA GEMM + GN partials ----------------
#define ASTRIDE 68   // dwords per LDS A row (64 used + 4 pad -> conflict-free b128 reads)
#define ABUF    (64 * ASTRIDE)

__launch_bounds__(256)
__global__ void k_main(const float* __restrict__ xt, const uint32_t* __restrict__ bp,
                       const float* __restrict__ dxp, const float* __restrict__ dyp,
                       const float* __restrict__ mp,
                       float* __restrict__ yout, float* __restrict__ stats) {
    __shared__ uint32_t ldsA[2 * ABUF];            // double-buffered 64px x 128 bf16 tile

    const int tid  = threadIdx.x;
    const int lane = tid & 31;
    const int wid  = tid >> 5;
    const int wr   = wid & 3;                      // pixel-row group (16 px)
    const int wc   = wid >> 2;                     // Cout half (64 ch)
    const int pix0 = blockIdx.x * 64;
    const int n    = pix0 / HW;                    // whole tile in one image (9216%64==0)

    // sampling role: 4 threads per pixel, 32 channels each
    const int pl  = tid >> 2;                      // local pixel 0..63
    const int cg  = tid & 3;                       // channel group of 32
    const int pix = pix0 + pl;
    const int hw  = pix % HW;
    const int h   = hw / Ww, w = hw % Ww;
    const float dx  = dxp[pix];
    const float dy  = dyp[pix];
    const float msk = mp[pix];

    v8f acc[4] = {};

    for (int t = 0; t < 9; ++t) {
        uint32_t* buf = &ldsA[(t & 1) * ABUF];
        // ---- bilinear sample 32 channels into LDS A tile (bf16) ----
        {
            float py = (float)(h - 1 + t / 3) + c_oyc[t] * dx;
            float px = (float)(w - 1 + t % 3) + c_oxc[t] * dy;
            float fy0 = floorf(py), fx0 = floorf(px);
            int   iy0 = (int)fy0,   ix0 = (int)fx0;
            int   iy1 = iy0 + 1,    ix1 = ix0 + 1;
            float fy = py - fy0, fx = px - fx0;
            float w00 = (1.f - fy) * (1.f - fx) * msk;
            float w01 = (1.f - fy) * fx * msk;
            float w10 = fy * (1.f - fx) * msk;
            float w11 = fy * fx * msk;
            bool vy0 = (iy0 >= 0) & (iy0 < Hh), vy1 = (iy1 >= 0) & (iy1 < Hh);
            bool vx0 = (ix0 >= 0) & (ix0 < Ww), vx1 = (ix1 >= 0) & (ix1 < Ww);
            if (!(vy0 && vx0)) w00 = 0.f;
            if (!(vy0 && vx1)) w01 = 0.f;
            if (!(vy1 && vx0)) w10 = 0.f;
            if (!(vy1 && vx1)) w11 = 0.f;
            int cy0 = iclamp(iy0, 0, Hh - 1), cy1 = iclamp(iy1, 0, Hh - 1);
            int cx0 = iclamp(ix0, 0, Ww - 1), cx1 = iclamp(ix1, 0, Ww - 1);
            const size_t nb = (size_t)n * HW;
            const float4* p00 = (const float4*)(xt + (nb + (size_t)cy0 * Ww + cx0) * Cin + cg * 32);
            const float4* p01 = (const float4*)(xt + (nb + (size_t)cy0 * Ww + cx1) * Cin + cg * 32);
            const float4* p10 = (const float4*)(xt + (nb + (size_t)cy1 * Ww + cx0) * Cin + cg * 32);
            const float4* p11 = (const float4*)(xt + (nb + (size_t)cy1 * Ww + cx1) * Cin + cg * 32);
            uint32_t* dst = &buf[pl * ASTRIDE + cg * 16];
            #pragma unroll
            for (int q = 0; q < 8; ++q) {
                float4 a = p00[q], b = p01[q], c = p10[q], d = p11[q];
                float r0 = w00 * a.x + w01 * b.x + w10 * c.x + w11 * d.x;
                float r1 = w00 * a.y + w01 * b.y + w10 * c.y + w11 * d.y;
                float r2 = w00 * a.z + w01 * b.z + w10 * c.z + w11 * d.z;
                float r3 = w00 * a.w + w01 * b.w + w10 * c.w + w11 * d.w;
                dst[q * 2]     = pack_bf16x2(r0, r1);
                dst[q * 2 + 1] = pack_bf16x2(r2, r3);
            }
        }
        __syncthreads();   // writes of buf done; prior-tap reads of the other buffer
                           // are ordered by the previous iteration's barrier

        // ---- 4 K-steps of 32 over this tap's 128 channels ----
        const int arow  = wr * 16 + (lane & 15);
        const int ahalf = lane >> 4;
        #pragma unroll
        for (int ks = 0; ks < 4; ++ks) {
            union { uint4 q[2]; v16bf v; } A;
            const uint32_t* ap = &buf[arow * ASTRIDE + ks * 16 + ahalf * 8];
            A.q[0] = *(const uint4*)(ap);
            A.q[1] = *(const uint4*)(ap + 4);
            const int kcg = t * 4 + ks;
            #pragma unroll
            for (int j = 0; j < 4; ++j) {
                union { uint4 q[2]; v16bf v; } B;
                const uint32_t* bq = bp + ((size_t)(kcg * 8 + (wc * 4 + j)) * 32 + lane) * 8;
                B.q[0] = *(const uint4*)(bq);
                B.q[1] = *(const uint4*)(bq + 4);
                acc[j] = __builtin_amdgcn_wmma_f32_16x16x32_bf16(
                    false, A.v, false, B.v, (short)0, acc[j], false, false);
            }
        }
    }

    // ---- write y (pix-major) + GN partial sums (one accumulator == one group) ----
    const int chalf = lane >> 4;
    const int ccol  = lane & 15;
    #pragma unroll
    for (int j = 0; j < 4; ++j) {
        float s = 0.f, s2 = 0.f;
        #pragma unroll
        for (int v = 0; v < 8; ++v) {
            float val = acc[j][v];
            int prow = pix0 + wr * 16 + v + 8 * chalf;
            yout[(size_t)prow * COUT + wc * 64 + j * 16 + ccol] = val;
            s  += val;
            s2 += val * val;
        }
        for (int m = 16; m > 0; m >>= 1) {
            s  += __shfl_xor(s,  m, 32);
            s2 += __shfl_xor(s2, m, 32);
        }
        if (lane == 0) {
            int g = wc * 4 + j;
            atomicAdd(&stats[n * GN_G + g], s);
            atomicAdd(&stats[32 + n * GN_G + g], s2);
        }
    }
}

// ---------------- kernel 5: finalize mu / rsigma ----------------
__global__ void k_fin(float* __restrict__ stats) {
    int i = threadIdx.x;                           // 0..31 == (n,g)
    float cnt = (float)(GPC * HW);
    float mu  = stats[i] / cnt;
    float var = stats[32 + i] / cnt - mu * mu;
    stats[64 + i] = mu;
    stats[96 + i] = rsqrtf(var + 1e-5f);
}

// ---------------- kernel 6: normalize + affine + relu -> NCHW out ----------------
__global__ void k_norm(const float* __restrict__ y, const float* __restrict__ stats,
                       const float* __restrict__ gamma, const float* __restrict__ beta,
                       float* __restrict__ out) {
    int pix = blockIdx.x;
    int c   = threadIdx.x;
    int n   = pix / HW;
    int hw  = pix % HW;
    int g   = c >> 4;
    float mu = stats[64 + n * GN_G + g];
    float rs = stats[96 + n * GN_G + g];
    float v  = (y[(size_t)pix * COUT + c] - mu) * rs * gamma[c] + beta[c];
    out[((size_t)n * COUT + c) * HW + hw] = fmaxf(v, 0.f);
}

extern "C" void kernel_launch(void* const* d_in, const int* in_sizes, int n_in,
                              void* d_out, int out_size, void* d_ws, size_t ws_size,
                              hipStream_t stream) {
    const float* x     = (const float*)d_in[0];   // 4*128*96*96
    const float* whead = (const float*)d_in[1];   // 3*128*3*3
    const float* bhead = (const float*)d_in[2];   // 3
    const float* wdcn  = (const float*)d_in[3];   // 128*128*3*3
    const float* gamma = (const float*)d_in[4];   // 128
    const float* beta  = (const float*)d_in[5];   // 128
    float* out = (float*)d_out;

    float*    ws   = (float*)d_ws;
    float*    xt   = ws + XT_OFF;
    float*    yb   = ws + Y_OFF;
    float*    dxp  = ws + DX_OFF;
    float*    dyp  = ws + DY_OFF;
    float*    mp   = ws + MK_OFF;
    float*    st   = ws + ST_OFF;
    uint32_t* bpck = (uint32_t*)(ws + BP_OFF);

    k_zero<<<1, 128, 0, stream>>>(st);
    k_transpose<<<NPIX, 128, 0, stream>>>(x, xt);
    k_packb<<<288, 256, 0, stream>>>(wdcn, bpck);          // 73728 dwords
    k_head<<<NPIX / 128, 128, 0, stream>>>(xt, whead, bhead, dxp, dyp, mp);
    k_main<<<NPIX / 64, 256, 0, stream>>>(xt, bpck, dxp, dyp, mp, yb, st);
    k_fin<<<1, 32, 0, stream>>>(st);
    k_norm<<<NPIX, 128, 0, stream>>>(yb, st, gamma, beta, out);
}
